// CausalSelfAttention_52063593562315
// MI455X (gfx1250) — compile-verified, matrix operand order checked by reading, not run on hardware
//
#include <hip/hip_runtime.h>
#include <hip/hip_bf16.h>

// ---------------------------------------------------------------------------
// CDNA5 (gfx1250) causal self-attention: bf16 WMMA GEMMs + flash attention,
// with Tensor-Data-Mover (TDM) tile staging into LDS.
// B=4, T=2048, C=1024, H=16, D=64.
// ---------------------------------------------------------------------------

typedef __attribute__((ext_vector_type(16))) __bf16 v16bf;
typedef __attribute__((ext_vector_type(8)))  float  v8f;
typedef __attribute__((ext_vector_type(8)))  __bf16 bf16x8;
typedef __attribute__((ext_vector_type(4)))  __bf16 bf16x4;

#define WMMA_BF16(A, B, C) \
    __builtin_amdgcn_wmma_f32_16x16x32_bf16(false, (A), false, (B), (short)0, (C), false, false)

// ---------------------------------------------------------------------------
// TDM: wave-level DMA of a 2-D bf16 tile from global to LDS (TENSORcnt).
// D# layout per CDNA5 ISA ch.8.  Uniform (SGPR) descriptors, EXEC ignored.
// This toolchain exposes the 6-arg builtin:
//   (u32x4 g0, i32x8 g1, i32x4 g2, i32x4 g3, i32x8 extra, int cpol)
// ---------------------------------------------------------------------------
#if __has_builtin(__builtin_amdgcn_tensor_load_to_lds) && \
    __has_builtin(__builtin_amdgcn_s_wait_tensorcnt)
#define USE_TDM 1
#else
#define USE_TDM 0
#endif

#if USE_TDM
typedef __attribute__((ext_vector_type(4))) unsigned int u32x4;
typedef __attribute__((ext_vector_type(8))) int          i32x8;
typedef __attribute__((ext_vector_type(4))) int          i32x4;

// tile_w elements per row, tile_h rows, global row stride in elements.
// pad_interval: pad after (2<<v) DWORDs; pad_amount: (v+1) DWORDs of pad.
static __device__ inline void tdm_load_2d(const void* gsrc, void* ldst,
                                          unsigned tile_w, unsigned tile_h,
                                          unsigned long long row_stride,
                                          unsigned pad_interval, unsigned pad_amount) {
    const unsigned long long ga = (unsigned long long)gsrc;
    const unsigned lds = (unsigned)(unsigned long long)ldst;  // LDS byte address

    u32x4 g0;
    g0[0] = 1u;                                   // count=1 (valid), is_restore=0
    g0[1] = lds;                                  // lds_addr[31:0]
    g0[2] = (unsigned)(ga & 0xffffffffull);       // global_addr[31:0]
    g0[3] = (unsigned)(ga >> 32) | (2u << 30);    // global_addr[56:32] | type=2

    const unsigned long long s0 = row_stride;     // tensor_dim0_stride (elements)
    i32x8 g1;
    g1[0] = (int)((1u << 16) |                    // data_size=1 -> 2 bytes
                  (1u << 20) |                    // pad_enable
                  (pad_interval << 22) | (pad_amount << 25));
    g1[1] = (int)((tile_w & 0xffffu) << 16);      // tensor_dim0[15:0] = tile_w
    g1[2] = (int)((tile_h & 0xffffu) << 16);      // tensor_dim0[31:16]=0 | tensor_dim1[15:0]
    g1[3] = (int)((tile_w & 0xffffu) << 16);      // tensor_dim1[31:16]=0 | tile_dim0
    g1[4] = (int)(tile_h & 0xffffu);              // tile_dim1 | tile_dim2=0
    g1[5] = (int)(unsigned)(s0 & 0xffffffffull);  // tensor_dim0_stride[31:0]
    g1[6] = (int)(unsigned)(s0 >> 32);            // stride0[47:32] | stride1[15:0]=0
    g1[7] = 0;

    i32x4 gz;
    gz[0] = 0; gz[1] = 0; gz[2] = 0; gz[3] = 0;   // 2-D tensor: groups 2/3 unused
    i32x8 gz8;
#pragma unroll
    for (int i = 0; i < 8; ++i) gz8[i] = 0;       // extra group (unused)

    __builtin_amdgcn_tensor_load_to_lds(g0, g1, gz, gz, gz8, 0);
}
static __device__ inline void tdm_wait0() { __builtin_amdgcn_s_wait_tensorcnt(0); }
#endif

// Load one 16-lane-striped bf16 fragment (A or B operand of 16x16x32).
// Per the CDNA5 16-bit operand layout, fragment elements 0..7 are
// k = 8*half .. 8*half+7 and elements 8..15 are k = 16+8*half .., i.e. two
// contiguous 16-byte reads (ds_read_b128 / global_load_b128).
static __device__ inline v16bf frag_ld(const __bf16* p, int half) {
    bf16x8 lo = *(const bf16x8*)(p + half * 8);
    bf16x8 hi = *(const bf16x8*)(p + 16 + half * 8);
    v16bf f;
#pragma unroll
    for (int i = 0; i < 8; ++i) { f[i] = lo[i]; f[i + 8] = hi[i]; }
    return f;
}

// ---------------------------------------------------------------------------
// fp32 -> bf16 convert (vectorized: 4 elements / thread)
// ---------------------------------------------------------------------------
__global__ __launch_bounds__(256) void cvt_bf16_kernel(const float* __restrict__ src,
                                                       __bf16* __restrict__ dst) {
    size_t i = ((size_t)blockIdx.x * 256 + threadIdx.x) * 4;
    const float4 v = *(const float4*)(src + i);
    bf16x4 o;
    o[0] = (__bf16)v.x; o[1] = (__bf16)v.y; o[2] = (__bf16)v.z; o[3] = (__bf16)v.w;
    *(bf16x4*)(dst + i) = o;
}

// ---------------------------------------------------------------------------
// Generic bf16 GEMM:  C[M,N] = A[M,K] * B[N,K]^T (+ bias)
// Block tile 128(M) x 256(N), 256 threads = 8 waves (2x4), wave tile 64x64:
// 16 accumulators -> 16 wmma per 32-wide k-step, 8 fragment loads.
// MODE 0: epilogue adds bias, scatters Q/K [B,H,T,D] and V transposed [B,H,D,T]
//         (branchless: base pointer + stride multipliers selected via cndmask).
// MODE 1: epilogue writes f32 (+ bias) to out[M,N].
// ---------------------------------------------------------------------------
template <int MODE>
__global__ __launch_bounds__(256) void gemm_bf16_kernel(
    const __bf16* __restrict__ A, const __bf16* __restrict__ B,
    const float* __restrict__ bias, float* __restrict__ outF,
    __bf16* __restrict__ qb, __bf16* __restrict__ kb, __bf16* __restrict__ vb,
    int M, int N, int K) {
    constexpr int LDT = 40;                 // padded LDS row stride (64B data + 16B pad)
    __shared__ __bf16 As[128 * LDT];        // 10 KB
    __shared__ __bf16 Bs[256 * LDT];        // 20 KB

    const int tid  = threadIdx.x;
    const int wave = tid >> 5, lane = tid & 31;
    const int half = lane >> 4, ln = lane & 15;
    const int wm = wave >> 2, wn = wave & 3;      // 2x4 wave grid
    const int m0 = blockIdx.y * 128, n0 = blockIdx.x * 256;

    v8f acc[4][4];
#pragma unroll
    for (int mi = 0; mi < 4; ++mi)
#pragma unroll
        for (int ni = 0; ni < 4; ++ni) acc[mi][ni] = (v8f){};

    const int arow = tid >> 1;              // 0..127
    const int aseg = (tid & 1) * 16;        // 0 / 16
    (void)arow; (void)aseg;

    for (int k0 = 0; k0 < K; k0 += 32) {
        __syncthreads();
#if USE_TDM
        if (wave == 0) {
            // 32x128 and 32x256 bf16 tiles; pad 4 DWORDs after every 16 DWORDs
            tdm_load_2d(A + (size_t)m0 * K + k0, As, 32, 128, (unsigned)K, 3, 3);
            tdm_load_2d(B + (size_t)n0 * K + k0, Bs, 32, 256, (unsigned)K, 3, 3);
            tdm_wait0();
        }
#else
        {   // stage A (2 x 16B) and B (4 x 16B) per thread
            const __bf16* ga = A + (size_t)(m0 + arow) * K + k0 + aseg;
            __bf16*       la = As + arow * LDT + aseg;
            ((bf16x8*)la)[0] = ((const bf16x8*)ga)[0];
            ((bf16x8*)la)[1] = ((const bf16x8*)ga)[1];
            const __bf16* gb = B + (size_t)(n0 + tid) * K + k0;
            __bf16*       lb = Bs + tid * LDT;
#pragma unroll
            for (int j = 0; j < 4; ++j) ((bf16x8*)lb)[j] = ((const bf16x8*)gb)[j];
        }
#endif
        __syncthreads();

        v16bf af[4], bfr[4];
#pragma unroll
        for (int mi = 0; mi < 4; ++mi)
            af[mi] = frag_ld(As + (wm * 64 + mi * 16 + ln) * LDT, half);
#pragma unroll
        for (int ni = 0; ni < 4; ++ni)
            bfr[ni] = frag_ld(Bs + (wn * 64 + ni * 16 + ln) * LDT, half);
#pragma unroll
        for (int mi = 0; mi < 4; ++mi)
#pragma unroll
            for (int ni = 0; ni < 4; ++ni)
                acc[mi][ni] = WMMA_BF16(af[mi], bfr[ni], acc[mi][ni]);
    }

    // Epilogue.  D layout: element r of lane L => m = r + 8*(L/16), n = L%16.
    if (MODE == 0) {
        const int b  = m0 >> 11;             // 128-row tile lies in one batch
        const int tb = m0 & 2047;            // t of tile row 0
#pragma unroll
        for (int mi = 0; mi < 4; ++mi) {
            const int tBase = tb + wm * 64 + mi * 16 + half * 8;
#pragma unroll
            for (int ni = 0; ni < 4; ++ni) {
                const int ng = n0 + wn * 64 + ni * 16 + ln;
                const float bv = bias[ng];
                const int which = ng >> 10;          // 0=q 1=k 2=v
                const int rem = ng & 1023;
                const int h = rem >> 6, d = rem & 63;
                // Q/K: [B,H,T,D] -> idx = bh*131072 + t*64 + d
                // V  : [B,H,D,T] -> idx = bh*131072 + d*2048 + t
                __bf16* bp = (which == 0) ? qb : ((which == 1) ? kb : vb);
                const int tmul = (which == 2) ? 1 : 64;
                const int dmul = (which == 2) ? 2048 : 1;
                const size_t basei =
                    (size_t)(b * 16 + h) * 131072 + (size_t)(d * dmul);
#pragma unroll
                for (int r = 0; r < 8; ++r)
                    bp[basei + (size_t)((tBase + r) * tmul)] =
                        (__bf16)(acc[mi][ni][r] + bv);
            }
        }
    } else {
#pragma unroll
        for (int mi = 0; mi < 4; ++mi) {
            const int mgBase = m0 + wm * 64 + mi * 16 + half * 8;
#pragma unroll
            for (int ni = 0; ni < 4; ++ni) {
                const int ng = n0 + wn * 64 + ni * 16 + ln;
                const float bv = bias[ng];
#pragma unroll
                for (int r = 0; r < 8; ++r)
                    outF[(size_t)(mgBase + r) * N + ng] = acc[mi][ni][r] + bv;
            }
        }
    }
}

// ---------------------------------------------------------------------------
// Flash attention (causal), one (64-query x head) block, 4 waves x 16 queries.
// Q/K bf16 [B*H, T, 64]; V bf16 transposed [B*H, 64, T].
// Writes y bf16 in [B, T, C] for the out-proj GEMM.
// ---------------------------------------------------------------------------
__global__ __launch_bounds__(128) void flash_attn_kernel(
    const __bf16* __restrict__ qb, const __bf16* __restrict__ kb,
    const __bf16* __restrict__ vb, __bf16* __restrict__ yb) {
    constexpr int KS = 72;                   // padded strides (128B data + 16B pad)
    constexpr int VS = 72;
    __shared__ __bf16 Ksh[64 * KS];          // K tile    [key][d]
    __shared__ __bf16 Vts[64 * VS];          // V^T tile  [d][key]
    __shared__ __bf16 Psh[4 * 16 * 64];      // per-wave P strip [16][64]

    const int tid  = threadIdx.x;
    const int wave = tid >> 5, lane = tid & 31;
    const int half = lane >> 4, ln = lane & 15;
    const int qt = blockIdx.x;               // query tile (64 rows)
    const int bh = blockIdx.y;               // b*16 + h
    const int b = bh >> 4, h = bh & 15;

    const size_t headQK = (size_t)bh * 2048 * 64;

    // Q fragments for the wave's 16-row strip (A layout: m = lane%16).
    v16bf qf[2];
    {
        const __bf16* qr = qb + headQK + (size_t)(qt * 64 + wave * 16 + ln) * 64;
#pragma unroll
        for (int s = 0; s < 2; ++s) qf[s] = frag_ld(qr + s * 32, half);
    }

    v8f o[4];
#pragma unroll
    for (int nt = 0; nt < 4; ++nt) o[nt] = (v8f){};
    float mrun[8], lrun[8];
#pragma unroll
    for (int r = 0; r < 8; ++r) { mrun[r] = -1e30f; lrun[r] = 0.0f; }

    __bf16* Pw = Psh + wave * 16 * 64;
    const float scale = 0.125f;              // 1/sqrt(64)

    const int srow = tid & 63;               // staging row
    const int scb  = (tid >> 6) * 32;        // staging 32-col half
    (void)srow; (void)scb;

    for (int kt = 0; kt <= qt; ++kt) {
        __syncthreads();
#if USE_TDM
        if (wave == 0) {
            // 64x64 bf16 tiles; pad 4 DWORDs after every 32 DWORDs
            tdm_load_2d(kb + headQK + (size_t)kt * 64 * 64, Ksh, 64, 64, 64u, 4, 3);
            tdm_load_2d(vb + (size_t)bh * 64 * 2048 + kt * 64, Vts, 64, 64, 2048u, 4, 3);
            tdm_wait0();
        }
#else
        {   // stage K [key][d] and V^T [d][key] tiles, b128 all the way
            const __bf16* ksrc = kb + headQK + (size_t)(kt * 64 + srow) * 64 + scb;
            __bf16*       kdst = Ksh + srow * KS + scb;
            const __bf16* vsrc = vb + ((size_t)bh * 64 + srow) * 2048 + kt * 64 + scb;
            __bf16*       vdst = Vts + srow * VS + scb;
#pragma unroll
            for (int j = 0; j < 4; ++j) {
                ((bf16x8*)kdst)[j] = ((const bf16x8*)ksrc)[j];
                ((bf16x8*)vdst)[j] = ((const bf16x8*)vsrc)[j];
            }
        }
#endif
        __syncthreads();

        // S = Q * K^T (B layout: n = key = lane%16, k = d contiguous)
        v8f s[4];
#pragma unroll
        for (int nt = 0; nt < 4; ++nt) {
            s[nt] = (v8f){};
#pragma unroll
            for (int ks = 0; ks < 2; ++ks) {
                v16bf kf = frag_ld(Ksh + (nt * 16 + ln) * KS + ks * 32, half);
                s[nt] = WMMA_BF16(qf[ks], kf, s[nt]);
            }
        }

        // scale + causal mask (only the diagonal tile needs per-element mask)
        if (kt == qt) {
#pragma unroll
            for (int nt = 0; nt < 4; ++nt)
#pragma unroll
                for (int r = 0; r < 8; ++r) {
                    const int n = kt * 64 + nt * 16 + ln;
                    const int m = qt * 64 + wave * 16 + r + half * 8;
                    s[nt][r] = (n <= m) ? s[nt][r] * scale : -1e30f;
                }
        } else {
#pragma unroll
            for (int nt = 0; nt < 4; ++nt)
#pragma unroll
                for (int r = 0; r < 8; ++r) s[nt][r] *= scale;
        }

        // online softmax: row max / sum via 16-lane xor-shuffle reductions
        float rmax[8];
#pragma unroll
        for (int r = 0; r < 8; ++r) {
            float v = s[0][r];
#pragma unroll
            for (int nt = 1; nt < 4; ++nt) v = fmaxf(v, s[nt][r]);
#pragma unroll
            for (int off = 1; off < 16; off <<= 1) v = fmaxf(v, __shfl_xor(v, off, 32));
            rmax[r] = v;
        }
        float alpha[8];
#pragma unroll
        for (int r = 0; r < 8; ++r) {
            const float mn = fmaxf(mrun[r], rmax[r]);
            alpha[r] = __expf(mrun[r] - mn);
            mrun[r] = mn;
        }
        float rsum[8];
#pragma unroll
        for (int r = 0; r < 8; ++r) rsum[r] = 0.0f;
#pragma unroll
        for (int nt = 0; nt < 4; ++nt)
#pragma unroll
            for (int r = 0; r < 8; ++r) {
                const float p = __expf(s[nt][r] - mrun[r]);
                s[nt][r] = p;
                rsum[r] += p;
            }
#pragma unroll
        for (int r = 0; r < 8; ++r) {
            float v = rsum[r];
#pragma unroll
            for (int off = 1; off < 16; off <<= 1) v += __shfl_xor(v, off, 32);
            lrun[r] = lrun[r] * alpha[r] + v;
        }
#pragma unroll
        for (int nt = 0; nt < 4; ++nt)
#pragma unroll
            for (int r = 0; r < 8; ++r) o[nt][r] *= alpha[r];

        // D-layout -> A-layout conversion for P via per-wave LDS round trip
#pragma unroll
        for (int nt = 0; nt < 4; ++nt)
#pragma unroll
            for (int r = 0; r < 8; ++r)
                Pw[(r + half * 8) * 64 + nt * 16 + ln] = (__bf16)s[nt][r];

        // O += P * V   (V^T rows are d, columns are contiguous keys)
#pragma unroll
        for (int ks = 0; ks < 2; ++ks) {
            v16bf pf = frag_ld(Pw + ln * 64 + ks * 32, half);
#pragma unroll
            for (int nt = 0; nt < 4; ++nt) {
                v16bf vf = frag_ld(Vts + (nt * 16 + ln) * VS + ks * 32, half);
                o[nt] = WMMA_BF16(pf, vf, o[nt]);
            }
        }
    }

    // normalize and write y[b][t][h*64+d] as bf16
    float inv[8];
#pragma unroll
    for (int r = 0; r < 8; ++r) inv[r] = 1.0f / lrun[r];
#pragma unroll
    for (int nt = 0; nt < 4; ++nt)
#pragma unroll
        for (int r = 0; r < 8; ++r) {
            const int t = qt * 64 + wave * 16 + r + half * 8;
            const int d = nt * 16 + ln;
            yb[((size_t)(b * 2048 + t)) * 1024 + h * 64 + d] = (__bf16)(o[nt][r] * inv[r]);
        }
}

// ---------------------------------------------------------------------------
// Host launcher
// ---------------------------------------------------------------------------
extern "C" void kernel_launch(void* const* d_in, const int* in_sizes, int n_in,
                              void* d_out, int out_size, void* d_ws, size_t ws_size,
                              hipStream_t stream) {
    constexpr int B = 4, T = 2048, C = 1024;
    constexpr int M = B * T;          // 8192
    constexpr int F = 3 * C;          // 3072

    const float* x      = (const float*)d_in[0];
    const float* w_attn = (const float*)d_in[1];
    const float* b_attn = (const float*)d_in[2];
    const float* w_proj = (const float*)d_in[3];
    const float* b_proj = (const float*)d_in[4];
    float* out = (float*)d_out;

    char* ws = (char*)d_ws;
    size_t off = 0;
    __bf16* xb  = (__bf16*)(ws + off); off += (size_t)M * C * 2;   // 16 MB
    __bf16* wab = (__bf16*)(ws + off); off += (size_t)F * C * 2;   //  6 MB
    __bf16* wpb = (__bf16*)(ws + off); off += (size_t)C * C * 2;   //  2 MB
    __bf16* qbf = (__bf16*)(ws + off); off += (size_t)M * C * 2;   // 16 MB
    __bf16* kbf = (__bf16*)(ws + off); off += (size_t)M * C * 2;   // 16 MB
    __bf16* vbf = (__bf16*)(ws + off); off += (size_t)M * C * 2;   // 16 MB (transposed)
    __bf16* ybf = (__bf16*)(ws + off); off += (size_t)M * C * 2;   // 16 MB

    // 1) fp32 -> bf16
    cvt_bf16_kernel<<<(size_t)M * C / 1024, 256, 0, stream>>>(x, xb);
    cvt_bf16_kernel<<<(size_t)F * C / 1024, 256, 0, stream>>>(w_attn, wab);
    cvt_bf16_kernel<<<(size_t)C * C / 1024, 256, 0, stream>>>(w_proj, wpb);

    // 2) QKV projection -> q/k [B,H,T,D], v transposed [B,H,D,T]
    gemm_bf16_kernel<0><<<dim3(F / 256, M / 128), 256, 0, stream>>>(
        xb, wab, b_attn, nullptr, qbf, kbf, vbf, M, F, C);

    // 3) causal flash attention
    flash_attn_kernel<<<dim3(T / 64, B * 16), 128, 0, stream>>>(qbf, kbf, vbf, ybf);

    // 4) output projection (f32 + bias) into d_out
    gemm_bf16_kernel<1><<<dim3(C / 256, M / 128), 256, 0, stream>>>(
        ybf, wpb, b_proj, out, nullptr, nullptr, nullptr, M, C, C);

    (void)in_sizes; (void)n_in; (void)out_size; (void)ws_size;
}
